// GatedGraphAttNet_46299747451282
// MI455X (gfx1250) — matrix operand order repaired and checked by reading, not compiled, not run on hardware
//
#include <hip/hip_runtime.h>
#include <hip/hip_bf16.h>

typedef __bf16 bf16_t;
typedef __attribute__((ext_vector_type(16))) __bf16 v16bf;
typedef __attribute__((ext_vector_type(8)))  __bf16 v8bf;
typedef __attribute__((ext_vector_type(8)))  float  v8f;
typedef __attribute__((ext_vector_type(4)))  __bf16 v4bf;

#define BLK_M 128
#define BLK_N 128
#define KSTEP 32   // one bf16 WMMA K-slice

#if __has_builtin(__builtin_amdgcn_global_load_async_to_lds_b128)
#define ASYNC_LDS 1
#else
#define ASYNC_LDS 0
#endif

// Exact parameter types of the async-copy builtin (from the compiler diagnostic):
// (int __vector(4) addrspace(1)*, int __vector(4) addrspace(3)*, imm int, imm int)
typedef int vec4i __attribute__((vector_size(16)));
typedef __attribute__((address_space(1))) vec4i* as1_v4i;
typedef __attribute__((address_space(3))) vec4i* as3_v4i;
#define GPTR(p) ((as1_v4i)(vec4i*)(p))
#define LPTR(p) ((as3_v4i)(vec4i*)(p))

__device__ __forceinline__ void async_wait0() {
#if __has_builtin(__builtin_amdgcn_s_wait_asynccnt)
  __builtin_amdgcn_s_wait_asynccnt(0);
#else
  asm volatile("s_wait_asynccnt 0x0" ::: "memory");
#endif
}

// ---------------------------------------------------------------------------
// Fragment loaders matching CDNA5 16-bit WMMA VGPR layouts (CDNA5 ISA)
// A (16x32, M x K): lane<16 -> row=lane, K chunks [0..7] and [16..23]
//                   lane>=16 -> row=lane-16, K chunks [8..15] and [24..31]
__device__ __forceinline__ v16bf load_frag_a(const bf16_t* base, int lane) {
  const int r  = lane & 15;
  const int kb = (lane < 16) ? 0 : 8;
  const bf16_t* p = base + r * KSTEP + kb;
  v8bf lo = *(const v8bf*)(p);
  v8bf hi = *(const v8bf*)(p + 16);
  return __builtin_shufflevector(lo, hi, 0,1,2,3,4,5,6,7,8,9,10,11,12,13,14,15);
}
// B (32x16, K x N) column-major per lane: lane%16 = column, lane<16 -> K chunks [0..7] and [16..23],
// lane>=16 -> K chunks [8..15] and [24..31] (same as A). Column n of W^T == row n of W,
// which is K-contiguous in the staging buffer.
__device__ __forceinline__ v16bf load_frag_b(const bf16_t* base, int lane) {
  const int n  = lane & 15;
  const int kb = (lane < 16) ? 0 : 8;
  const bf16_t* p = base + n * KSTEP + kb;
  v8bf lo = *(const v8bf*)(p);
  v8bf hi = *(const v8bf*)(p + 16);
  return __builtin_shufflevector(lo, hi, 0,1,2,3,4,5,6,7,8,9,10,11,12,13,14,15);
}

// ---------------------------------------------------------------------------
// Pipelined NT GEMM:  C[M x N] (+)= A[M x K] * B[N x K]^T   (bf16 in, f32 out)
// grid = (N/128, M/128, batch), block = 256 threads (8 waves, 4(M) x 2(N)).
// Each wave computes 32x64 = 8x v_wmma_f32_16x16x32_bf16 per K-slice.
// Ping-pong LDS; slice s+1 is staged (async DMA if available, else via VGPRs)
// while the WMMAs of slice s run; one barrier per K-slice.
template <bool ACCUM>
__global__ void __launch_bounds__(256)
gemm_nt(const bf16_t* __restrict__ A, int lda, long long sA,
        const bf16_t* __restrict__ B, int ldb, long long sB,
        float* __restrict__ C, int ldc, long long sC, int K)
{
  __shared__ __attribute__((aligned(64))) bf16_t lsA[2][BLK_M * KSTEP];
  __shared__ __attribute__((aligned(64))) bf16_t lsB[2][BLK_N * KSTEP];

  const int tid  = threadIdx.x;
  const int lane = tid & 31;
  const int wave = tid >> 5;
  const int wm   = (wave & 3) * 32;   // 4 waves along M
  const int wn   = (wave >> 2) * 64;  // 2 waves along N

  const long long zb = blockIdx.z;
  A += zb * sA + (long long)(blockIdx.y * BLK_M) * lda;
  B += zb * sB + (long long)(blockIdx.x * BLK_N) * ldb;
  C += zb * sC;

  // Branch-free staging: 128 rows x 4 segs = 512 chunks of 16B; 2 per thread.
  const int r0 = tid >> 2;            // rows r0 and r0+64
  const int s0 = (tid & 3) * 8;       // K-seg offset in elements

  v8f acc[2][4] = {};
  const int nsteps = K / KSTEP;

#if ASYNC_LDS
  // ---- direct global -> LDS async DMA staging (ASYNCcnt) ----
  {
    __builtin_amdgcn_global_load_async_to_lds_b128(
        GPTR(&A[(long long)(r0     ) * lda + s0]), LPTR(&lsA[0][(r0     ) * KSTEP + s0]), 0, 0);
    __builtin_amdgcn_global_load_async_to_lds_b128(
        GPTR(&A[(long long)(r0 + 64) * lda + s0]), LPTR(&lsA[0][(r0 + 64) * KSTEP + s0]), 0, 0);
    __builtin_amdgcn_global_load_async_to_lds_b128(
        GPTR(&B[(long long)(r0     ) * ldb + s0]), LPTR(&lsB[0][(r0     ) * KSTEP + s0]), 0, 0);
    __builtin_amdgcn_global_load_async_to_lds_b128(
        GPTR(&B[(long long)(r0 + 64) * ldb + s0]), LPTR(&lsB[0][(r0 + 64) * KSTEP + s0]), 0, 0);
    async_wait0();
  }
  __syncthreads();

  for (int step = 0; step < nsteps; ++step) {
    const int  cur  = step & 1;
    const bool more = (step + 1) < nsteps;

    if (more) {   // DMA next slice into the other buffer; overlaps with WMMAs below
      const int nxt = cur ^ 1;
      const long long ko = (long long)(step + 1) * KSTEP;
      __builtin_amdgcn_global_load_async_to_lds_b128(
          GPTR(&A[(long long)(r0     ) * lda + ko + s0]), LPTR(&lsA[nxt][(r0     ) * KSTEP + s0]), 0, 0);
      __builtin_amdgcn_global_load_async_to_lds_b128(
          GPTR(&A[(long long)(r0 + 64) * lda + ko + s0]), LPTR(&lsA[nxt][(r0 + 64) * KSTEP + s0]), 0, 0);
      __builtin_amdgcn_global_load_async_to_lds_b128(
          GPTR(&B[(long long)(r0     ) * ldb + ko + s0]), LPTR(&lsB[nxt][(r0     ) * KSTEP + s0]), 0, 0);
      __builtin_amdgcn_global_load_async_to_lds_b128(
          GPTR(&B[(long long)(r0 + 64) * ldb + ko + s0]), LPTR(&lsB[nxt][(r0 + 64) * KSTEP + s0]), 0, 0);
    }

    const v16bf a0 = load_frag_a(&lsA[cur][(wm +  0) * KSTEP], lane);
    const v16bf a1 = load_frag_a(&lsA[cur][(wm + 16) * KSTEP], lane);
    #pragma unroll
    for (int tn = 0; tn < 4; ++tn) {
      const v16bf b = load_frag_b(&lsB[cur][(wn + tn * 16) * KSTEP], lane);
      acc[0][tn] = __builtin_amdgcn_wmma_f32_16x16x32_bf16(false, a0, false, b, (short)0, acc[0][tn], false, false);
      acc[1][tn] = __builtin_amdgcn_wmma_f32_16x16x32_bf16(false, a1, false, b, (short)0, acc[1][tn], false, false);
    }

    if (more) {
      async_wait0();      // own DMA complete; barrier makes all waves' LDS visible
      __syncthreads();
    }
  }
#else
  // ---- register-staged fallback (proven path) ----
  uint4 ta0, ta1, tb0, tb1;
  ta0 = *(const uint4*)&A[(long long)(r0     ) * lda + s0];
  ta1 = *(const uint4*)&A[(long long)(r0 + 64) * lda + s0];
  tb0 = *(const uint4*)&B[(long long)(r0     ) * ldb + s0];
  tb1 = *(const uint4*)&B[(long long)(r0 + 64) * ldb + s0];
  *(uint4*)&lsA[0][(r0     ) * KSTEP + s0] = ta0;
  *(uint4*)&lsA[0][(r0 + 64) * KSTEP + s0] = ta1;
  *(uint4*)&lsB[0][(r0     ) * KSTEP + s0] = tb0;
  *(uint4*)&lsB[0][(r0 + 64) * KSTEP + s0] = tb1;
  __syncthreads();

  for (int step = 0; step < nsteps; ++step) {
    const int  cur  = step & 1;
    const bool more = (step + 1) < nsteps;

    if (more) {   // issue next-slice global loads early; latency hides behind WMMAs
      const long long ko = (long long)(step + 1) * KSTEP;
      ta0 = *(const uint4*)&A[(long long)(r0     ) * lda + ko + s0];
      ta1 = *(const uint4*)&A[(long long)(r0 + 64) * lda + ko + s0];
      tb0 = *(const uint4*)&B[(long long)(r0     ) * ldb + ko + s0];
      tb1 = *(const uint4*)&B[(long long)(r0 + 64) * ldb + ko + s0];
    }

    const v16bf a0 = load_frag_a(&lsA[cur][(wm +  0) * KSTEP], lane);
    const v16bf a1 = load_frag_a(&lsA[cur][(wm + 16) * KSTEP], lane);
    #pragma unroll
    for (int tn = 0; tn < 4; ++tn) {
      const v16bf b = load_frag_b(&lsB[cur][(wn + tn * 16) * KSTEP], lane);
      acc[0][tn] = __builtin_amdgcn_wmma_f32_16x16x32_bf16(false, a0, false, b, (short)0, acc[0][tn], false, false);
      acc[1][tn] = __builtin_amdgcn_wmma_f32_16x16x32_bf16(false, a1, false, b, (short)0, acc[1][tn], false, false);
    }

    if (more) {
      const int nxt = cur ^ 1;
      *(uint4*)&lsA[nxt][(r0     ) * KSTEP + s0] = ta0;
      *(uint4*)&lsA[nxt][(r0 + 64) * KSTEP + s0] = ta1;
      *(uint4*)&lsB[nxt][(r0     ) * KSTEP + s0] = tb0;
      *(uint4*)&lsB[nxt][(r0 + 64) * KSTEP + s0] = tb1;
      __syncthreads();
    }
  }
#endif

  // C/D layout: VGPR v -> M = v + (lane<16?0:8), N = lane%16
  const int cr = (lane < 16) ? 0 : 8;
  const int cn = lane & 15;
  #pragma unroll
  for (int tm = 0; tm < 2; ++tm) {
    #pragma unroll
    for (int tn = 0; tn < 4; ++tn) {
      const long long row = (long long)blockIdx.y * BLK_M + wm + tm * 16 + cr;
      const long long col = (long long)blockIdx.x * BLK_N + wn + tn * 16 + cn;
      #pragma unroll
      for (int v = 0; v < 8; ++v) {
        const long long off = (row + v) * (long long)ldc + col;
        if (ACCUM) C[off] += acc[tm][tn][v];
        else       C[off]  = acc[tm][tn][v];
      }
    }
  }
}

// ---------------------------------------------------------------------------
// Vectorized f32 -> bf16 conversion (4 elements per thread; n % 4 == 0).
__global__ void __launch_bounds__(256)
cvt_f32_bf16(const float* __restrict__ in, bf16_t* __restrict__ out, long long n4)
{
  const long long i = (long long)blockIdx.x * 256 + threadIdx.x;
  if (i >= n4) return;
  const float4 f = ((const float4*)in)[i];
  v4bf o; o.x = (bf16_t)f.x; o.y = (bf16_t)f.y; o.z = (bf16_t)f.z; o.w = (bf16_t)f.w;
  ((v4bf*)out)[i] = o;
}

// feat_proj f32 -> (+b_fc) -> fpT bf16 [b][d][n], plus q,k GEMVs. 1 thread per node.
__global__ void __launch_bounds__(256)
build_fpT_qk(const float* __restrict__ fp, const float* __restrict__ bfc,
             const float* __restrict__ wq, const float* __restrict__ bq,
             const float* __restrict__ wk, const float* __restrict__ bk,
             bf16_t* __restrict__ fpT, float* __restrict__ q, float* __restrict__ k)
{
  const int m = blockIdx.x * 256 + threadIdx.x;     // m = b*1024 + n, < 16384
  const int b = m >> 10, n = m & 1023;
  const float* fr = fp + (long long)m * 768;
  bf16_t* ft = fpT + (long long)b * 768 * 1024 + n;
  float qa = bq[0], ka = bk[0];
  for (int e = 0; e < 768; ++e) {
    const float v = fr[e] + bfc[e];
    qa += v * wq[e];
    ka += v * wk[e];
    ft[(long long)e * 1024] = (bf16_t)v;
  }
  q[m] = qa; k[m] = ka;
}

// One block per attention row: leaky_relu(masked q_i+k_j) -> softmax -> bf16
__global__ void __launch_bounds__(256)
att_softmax(const float* __restrict__ adj, const float* __restrict__ q,
            const float* __restrict__ k, bf16_t* __restrict__ att)
{
  __shared__ float vals[1024];
  __shared__ float red[256];
  const int row = blockIdx.x;              // b*1024 + i
  const int b   = row >> 10;
  const int tid = threadIdx.x;
  const float* adjr = adj + (long long)row * 1024;
  const float* kb   = k + b * 1024;
  const float  qv   = q[row];

  float lmax = -3.4e38f;
  for (int j = tid; j < 1024; j += 256) {
    float l = qv + kb[j] + (1.0f - adjr[j]) * (-1.0e9f);
    l = (l >= 0.0f) ? l : 0.01f * l;       // leaky_relu
    vals[j] = l;
    lmax = fmaxf(lmax, l);
  }
  red[tid] = lmax; __syncthreads();
  for (int s = 128; s > 0; s >>= 1) {
    if (tid < s) red[tid] = fmaxf(red[tid], red[tid + s]);
    __syncthreads();
  }
  const float m = red[0]; __syncthreads();

  float lsum = 0.0f;
  for (int j = tid; j < 1024; j += 256) {
    const float e = __expf(vals[j] - m);
    vals[j] = e; lsum += e;
  }
  red[tid] = lsum; __syncthreads();
  for (int s = 128; s > 0; s >>= 1) {
    if (tid < s) red[tid] += red[tid + s];
    __syncthreads();
  }
  const float inv = 1.0f / red[0];
  bf16_t* o = att + (long long)row * 1024;
  for (int j = tid; j < 1024; j += 256) o[j] = (bf16_t)(vals[j] * inv);
}

__device__ __forceinline__ float sigmoidf_(float x) { return 1.0f / (1.0f + __expf(-x)); }

// u = sigmoid(Uacc+b), written in-place; rx = bf16(sigmoid(Racc+b) * x)
__global__ void __launch_bounds__(256)
gate1(float* __restrict__ uacc, const float* __restrict__ racc,
      const float* __restrict__ x,
      const float* __restrict__ buy, const float* __restrict__ bux,
      const float* __restrict__ bry, const float* __restrict__ brx,
      bf16_t* __restrict__ rx, long long n)
{
  const long long i = (long long)blockIdx.x * 256 + threadIdx.x;
  if (i >= n) return;
  const int e = (int)(i % 768);
  const float u = sigmoidf_(uacc[i] + buy[e] + bux[e]);
  uacc[i] = u;
  const float r = sigmoidf_(racc[i] + bry[e] + brx[e]);
  rx[i] = (bf16_t)(r * x[i]);
}

// out = (1-u)*x + u*tanh(Tacc + b_ty + b_tx)
__global__ void __launch_bounds__(256)
gate2(const float* __restrict__ tacc, const float* __restrict__ u,
      const float* __restrict__ x,
      const float* __restrict__ bty, const float* __restrict__ btx,
      float* __restrict__ out, long long n)
{
  const long long i = (long long)blockIdx.x * 256 + threadIdx.x;
  if (i >= n) return;
  const int e = (int)(i % 768);
  const float xt = tanhf(tacc[i] + bty[e] + btx[e]);
  const float uu = u[i];
  out[i] = (1.0f - uu) * x[i] + uu * xt;
}

// ---------------------------------------------------------------------------
extern "C" void kernel_launch(void* const* d_in, const int* in_sizes, int n_in,
                              void* d_out, int out_size, void* d_ws, size_t ws_size,
                              hipStream_t stream)
{
  (void)in_sizes; (void)n_in; (void)out_size; (void)ws_size;
  const int Bn = 16, Nn = 1024, Dd = 768;
  const long long M  = (long long)Bn * Nn;   // 16384 rows
  const long long MD = M * Dd;               // 12,582,912 elems

  const float* x    = (const float*)d_in[0];
  const float* adj  = (const float*)d_in[1];
  const float* W_fc = (const float*)d_in[2];
  const float* b_fc = (const float*)d_in[3];
  const float* w_q  = (const float*)d_in[4];
  const float* b_q  = (const float*)d_in[5];
  const float* w_k  = (const float*)d_in[6];
  const float* b_k  = (const float*)d_in[7];
  const float* W_uy = (const float*)d_in[8];  const float* b_uy = (const float*)d_in[9];
  const float* W_ux = (const float*)d_in[10]; const float* b_ux = (const float*)d_in[11];
  const float* W_ry = (const float*)d_in[12]; const float* b_ry = (const float*)d_in[13];
  const float* W_rx = (const float*)d_in[14]; const float* b_rx = (const float*)d_in[15];
  const float* W_ty = (const float*)d_in[16]; const float* b_ty = (const float*)d_in[17];
  const float* W_tx = (const float*)d_in[18]; const float* b_tx = (const float*)d_in[19];
  float* out = (float*)d_out;

  // ---- workspace carve-out (~320 MB total) ----
  char* ws = (char*)d_ws;
  size_t off = 0;
  auto carve = [&](size_t bytes) -> void* {
    void* p = ws + off;
    off = (off + bytes + 255) & ~(size_t)255;
    return p;
  };
  bf16_t* Xbf  = (bf16_t*)carve((size_t)MD * 2);                 // inputs bf16
  bf16_t* Wbf  = (bf16_t*)carve((size_t)7 * Dd * Dd * 2);        // 7 weights bf16
  float*  FP   = (float*) carve((size_t)MD * 4);                 // feat_proj f32; reused as y f32
  bf16_t* FPT  = (bf16_t*)carve((size_t)Bn * Dd * Nn * 2);       // feat_proj^T bf16
  float*  Qv   = (float*) carve((size_t)M * 4);
  float*  Kv   = (float*) carve((size_t)M * 4);
  bf16_t* ATT  = (bf16_t*)carve((size_t)Bn * Nn * Nn * 2);       // att bf16; reused as r*x bf16
  bf16_t* Ybf  = (bf16_t*)carve((size_t)MD * 2);
  float*  Uacc = (float*) carve((size_t)MD * 4);
  float*  Racc = (float*) carve((size_t)MD * 4);
  float*  Tacc = (float*) carve((size_t)MD * 4);

  bf16_t* Wfc = Wbf + 0ll * Dd * Dd;
  bf16_t* Wuy = Wbf + 1ll * Dd * Dd;
  bf16_t* Wux = Wbf + 2ll * Dd * Dd;
  bf16_t* Wry = Wbf + 3ll * Dd * Dd;
  bf16_t* Wrx = Wbf + 4ll * Dd * Dd;
  bf16_t* Wty = Wbf + 5ll * Dd * Dd;
  bf16_t* Wtx = Wbf + 6ll * Dd * Dd;

  // ---- precision conversions (4-wide) ----
  cvt_f32_bf16<<<dim3((unsigned)((MD / 4 + 255) / 256)), 256, 0, stream>>>(x, Xbf, MD / 4);
  const float* Wsrc[7] = {W_fc, W_uy, W_ux, W_ry, W_rx, W_ty, W_tx};
  for (int i = 0; i < 7; ++i)
    cvt_f32_bf16<<<dim3((Dd * Dd / 4 + 255) / 256), 256, 0, stream>>>(
        Wsrc[i], Wbf + (long long)i * Dd * Dd, (long long)Dd * Dd / 4);

  // 1) feat_proj = X @ W_fc^T           (M x 768, K=768)
  dim3 gW(Dd / BLK_N, (unsigned)(M / BLK_M), 1);
  gemm_nt<false><<<gW, 256, 0, stream>>>(Xbf, Dd, 0, Wfc, Dd, 0, FP, Dd, 0, Dd);

  // 2) add bias, build fpT bf16 (transposed), q,k GEMVs
  build_fpT_qk<<<dim3((unsigned)(M / 256)), 256, 0, stream>>>(
      FP, b_fc, w_q, b_q, w_k, b_k, FPT, Qv, Kv);

  // 3) masked leaky softmax -> att bf16
  att_softmax<<<dim3((unsigned)M), 256, 0, stream>>>(adj, Qv, Kv, ATT);

  // 4) y[b] = att[b] @ fpT[b]^T   (batched, K=1024) -> reuse FP as y f32
  dim3 gY(Dd / BLK_N, Nn / BLK_M, Bn);
  gemm_nt<false><<<gY, 256, 0, stream>>>(ATT, Nn, (long long)Nn * Nn,
                                         FPT, Nn, (long long)Dd * Nn,
                                         FP, Dd, (long long)Nn * Dd, Nn);

  // 5) y -> bf16
  cvt_f32_bf16<<<dim3((unsigned)((MD / 4 + 255) / 256)), 256, 0, stream>>>(FP, Ybf, MD / 4);

  // 6) gating GEMMs (accumulate pairs into single f32 buffers)
  gemm_nt<false><<<gW, 256, 0, stream>>>(Ybf, Dd, 0, Wty, Dd, 0, Tacc, Dd, 0, Dd);
  gemm_nt<false><<<gW, 256, 0, stream>>>(Ybf, Dd, 0, Wuy, Dd, 0, Uacc, Dd, 0, Dd);
  gemm_nt<true ><<<gW, 256, 0, stream>>>(Xbf, Dd, 0, Wux, Dd, 0, Uacc, Dd, 0, Dd);
  gemm_nt<false><<<gW, 256, 0, stream>>>(Ybf, Dd, 0, Wry, Dd, 0, Racc, Dd, 0, Dd);
  gemm_nt<true ><<<gW, 256, 0, stream>>>(Xbf, Dd, 0, Wrx, Dd, 0, Racc, Dd, 0, Dd);

  // 7) u = sigmoid(.), rx = bf16(r * x)  (rx reuses ATT buffer, att is consumed)
  bf16_t* RX = ATT;
  gate1<<<dim3((unsigned)((MD + 255) / 256)), 256, 0, stream>>>(
      Uacc, Racc, x, b_uy, b_ux, b_ry, b_rx, RX, MD);

  // 8) Tacc += (r*x) @ W_tx^T
  gemm_nt<true><<<gW, 256, 0, stream>>>(RX, Dd, 0, Wtx, Dd, 0, Tacc, Dd, 0, Dd);

  // 9) out = (1-u)*x + u*tanh(Tacc + b)
  gate2<<<dim3((unsigned)((MD + 255) / 256)), 256, 0, stream>>>(
      Tacc, Uacc, x, b_ty, b_tx, out, MD);
}
